// LSTMModel_74801150427653
// MI455X (gfx1250) — matrix operand order repaired and checked by reading, not compiled, not run on hardware
//
#include <hip/hip_runtime.h>
#include <stdint.h>

// ---------------------------------------------------------------------------
// Fused 2-layer LSTM (B=4096, T=512, H=32) + MLP head for MI455X (gfx1250).
// Each wave32 owns 16 batch rows and runs the whole T=512 recurrence with
// V_WMMA_F32_16X16X32_F16 (K=32 == hidden size, so one WMMA per gate tile).
// All 24 weight B-tiles live in VGPRs for the whole recurrence; x is
// double-buffered in LDS and prefetched with GLOBAL_LOAD_ASYNC_TO_LDS_B32.
// ---------------------------------------------------------------------------

#define B_TOT   4096
#define T_TOT   512
#define HID     32
#define XCHUNK  32           // steps per x chunk (double buffered)
#define WPB     4            // waves per block
#define RPW     16           // batch rows per wave

#define USE_ASYNC_X 1        // gfx1250 async global->LDS prefetch of x

typedef _Float16 v16h __attribute__((ext_vector_type(16)));
typedef _Float16 v8h  __attribute__((ext_vector_type(8)));
typedef float    v8f  __attribute__((ext_vector_type(8)));
typedef float    v4f  __attribute__((ext_vector_type(4)));

// ---------------- activations: native CDNA5 v_tanh_f32 ---------------------
static __device__ __forceinline__ float tanh_fast(float x) {
#if __has_builtin(__builtin_amdgcn_tanhf)
  return __builtin_amdgcn_tanhf(x);
#else
  float r;
  asm("v_tanh_f32 %0, %1" : "=v"(r) : "v"(x));
  return r;
#endif
}
static __device__ __forceinline__ float sig_fast(float x) {
  // sigmoid(x) = 0.5*tanh(0.5x)+0.5  -> single transcendental per gate
  return __builtin_fmaf(0.5f, tanh_fast(0.5f * x), 0.5f);
}
static __device__ __forceinline__ v8f v8_sig(v8f x) {
  v8f r;
#pragma unroll
  for (int i = 0; i < 8; ++i) r[i] = sig_fast(x[i]);
  return r;
}
static __device__ __forceinline__ v8f v8_tanh(v8f x) {
  v8f r;
#pragma unroll
  for (int i = 0; i < 8; ++i) r[i] = tanh_fast(x[i]);
  return r;
}

// ---------------- WMMA helpers --------------------------------------------
static __device__ __forceinline__ v8f wmma16x16x32(v16h a, v16h b, v8f c) {
  return __builtin_amdgcn_wmma_f32_16x16x32_f16(
      /*neg_a=*/false, a, /*neg_b=*/false, b,
      /*c_mod=*/(short)0, c, /*reuse_a=*/false, /*reuse_b=*/false);
}
static __device__ __forceinline__ v16h cat8(v8h lo, v8h hi) {
  v16h a;
#pragma unroll
  for (int i = 0; i < 8; ++i) { a[i] = lo[i]; a[i + 8] = hi[i]; }
  return a;
}

// A operand (16x32 f16) from row-major LDS h[16][32]:
//  lanes 0-15 : M=lane,   K 0..7 in v0-3, K 16..23 in v4-7
//  lanes16-31 : M=lane-16,K 8..15 in v0-3, K 24..31 in v4-7
static __device__ __forceinline__ v16h loadA(const _Float16* base, int lane) {
  const int off = (lane & 15) * HID + (lane >> 4) * 8;   // in halves
  v8h lo = *(const v8h*)(base + off);                    // K 0..7  (+8*hi)
  v8h hi = *(const v8h*)(base + off + 16);               // K 16..23(+8*hi)
  return cat8(lo, hi);
}

// B operand tile (32x16 f16) pre-packed per lane: 16 contiguous halves.
static __device__ __forceinline__ v16h loadW(const _Float16* w, int mat,
                                             int j, int lane) {
  const _Float16* p = w + (((mat * 8) + j) * 32 + lane) * 16;
  v8h lo = *(const v8h*)p;
  v8h hi = *(const v8h*)(p + 8);
  return cat8(lo, hi);
}

// Store h (f32, D layout: lane holds col n, vgpr r holds row 8*(lane>=16)+r)
// into row-major f16 LDS h[16][32].
static __device__ __forceinline__ void storeH(_Float16* base, v8f h, int hh,
                                              int lane) {
  const int n  = 16 * hh + (lane & 15);
  const int m0 = (lane >> 4) * 8;
#pragma unroll
  for (int r = 0; r < 8; ++r) base[(m0 + r) * HID + n] = (_Float16)h[r];
}

// ---------------- async global->LDS (CDNA5, ASYNCcnt tracked) --------------
#if USE_ASYNC_X
static __device__ __forceinline__ void async_copy_b32(uint32_t lds_off,
                                                      uint64_t gaddr) {
  // per-lane: LDS[lds_off] = MEM[gaddr], 4 bytes, tracked by ASYNCcnt
  asm volatile("global_load_async_to_lds_b32 %0, %1, off"
               :: "v"(lds_off), "v"(gaddr));
}
static __device__ __forceinline__ void wait_asynccnt0() {
  asm volatile("s_wait_asynccnt 0x0" ::: "memory");
}
// Stage one x chunk [16 rows x XCHUNK t] transposed to LDS [t][row] layout.
// xg = &x[b0*T + tb]; s_dst = &s_x[wv][buf][0].
static __device__ __forceinline__ void stage_x_async(const float* xg,
                                                     float* s_dst, int lane) {
  const int row = lane & 15;
  const int th  = lane >> 4;                       // covers t-half of chunk
  uint32_t l0 = (uint32_t)(uintptr_t)s_dst +
                (uint32_t)(((th * 16) * RPW + row) * 4);
  uint64_t g0 = (uint64_t)(uintptr_t)(xg + (size_t)row * T_TOT + th * 16);
#pragma unroll
  for (int i = 0; i < 16; ++i)
    async_copy_b32(l0 + (uint32_t)i * (RPW * 4), g0 + (uint64_t)i * 4);
}
#else
static __device__ __forceinline__ void stage_x_sync(const float* xg,
                                                    float* s_dst, int lane) {
  const int row = lane & 15;
  const int th  = lane >> 4;
  const float* g = xg + (size_t)row * T_TOT + th * 16;
#pragma unroll
  for (int i = 0; i < 4; ++i) {
    v4f v = *(const v4f*)(g + 4 * i);
    const int tb = th * 16 + 4 * i;
#pragma unroll
    for (int e = 0; e < 4; ++e) s_dst[(tb + e) * RPW + row] = v[e];
  }
}
#endif

// ---------------------------------------------------------------------------
// Pack kernel: f32 weights -> f16 B-operand lane layout in d_ws.
// B tile j covers gate cols 16j..16j+15.  For lane L, element e (0..15):
//   K = 8*(L/16)+e (+8 if e>=8) , N = L%16 , value = W[16j+N][K]   (W is [128][32])
// ---------------------------------------------------------------------------
__global__ void lstm_pack_weights(const float* __restrict__ whh0,
                                  const float* __restrict__ wih1,
                                  const float* __restrict__ whh1,
                                  _Float16* __restrict__ pk) {
  int tid = blockIdx.x * blockDim.x + threadIdx.x;   // 0..767
  if (tid >= 3 * 8 * 32) return;
  const int mat = tid >> 8;
  const int j   = (tid >> 5) & 7;
  const int L   = tid & 31;
  const float* w = (mat == 0) ? whh0 : (mat == 1 ? wih1 : whh1);
  const int g     = 16 * j + (L & 15);
  const int kbase = 8 * (L >> 4);
  _Float16* dst = pk + ((mat * 8 + j) * 32 + L) * 16;
#pragma unroll
  for (int e = 0; e < 16; ++e) dst[e] = (_Float16)w[g * HID + kbase + e + ((e & 8) ? 8 : 0)];
}

// ---------------------------------------------------------------------------
// Main fused kernel. grid = 64 blocks x 128 threads (4 waves); wave -> 16 rows.
// ---------------------------------------------------------------------------
__global__ __launch_bounds__(128) void lstm_fused_kernel(
    const float* __restrict__ x,
    const float* __restrict__ wih0,
    const float* __restrict__ bih0, const float* __restrict__ bhh0,
    const float* __restrict__ bih1, const float* __restrict__ bhh1,
    const float* __restrict__ fc1w, const float* __restrict__ fc1b,
    const float* __restrict__ fc2w, const float* __restrict__ fc2b,
    const _Float16* __restrict__ pk,
    float* __restrict__ out) {
  __shared__ __align__(16) _Float16 s_h[WPB][2][RPW * HID];       // 8 KB
  __shared__ __align__(16) float    s_x[WPB][2][XCHUNK * RPW];    // 16 KB

  const int tid  = threadIdx.x;
  const int wv   = tid >> 5;
  const int lane = tid & 31;
  const int b0   = (blockIdx.x * WPB + wv) * RPW;

  // Weight B-tiles live in VGPRs for the whole recurrence (explicit SSA
  // values so the async-wait asm clobber can't force LDS reloads).
  v16h w0[8], w1[8], w2[8];
#pragma unroll
  for (int j = 0; j < 8; ++j) {
    w0[j] = loadW(pk, 0, j, lane);       // w_hh0 tiles
    w1[j] = loadW(pk, 1, j, lane);       // w_ih1 tiles
    w2[j] = loadW(pk, 2, j, lane);       // w_hh1 tiles
  }

  // Per-lane constants: gate col n = lane%16 of tile j -> gate g = 16j+n.
  float wih0s[8], bias0s[8], bias1s[8];
#pragma unroll
  for (int j = 0; j < 8; ++j) {
    const int g = 16 * j + (lane & 15);
    wih0s[j]  = wih0[g];                 // input_size == 1
    bias0s[j] = bih0[g] + bhh0[g];
    bias1s[j] = bih1[g] + bhh1[g];
  }

  // Cell state in WMMA C/D layout, f32: [half][8 rows-per-lane].
  v8f c0[2], c1[2];
#pragma unroll
  for (int h = 0; h < 2; ++h)
#pragma unroll
    for (int r = 0; r < 8; ++r) { c0[h][r] = 0.0f; c1[h][r] = 0.0f; }

  // Register-carried A operands (h(t-1) of each layer); zero at t=0, so the
  // LDS h buffers are written before they are ever read.
  v16h h0A, h1A;
#pragma unroll
  for (int i = 0; i < 16; ++i) {
    h0A[i] = (_Float16)0.0f;
    h1A[i] = (_Float16)0.0f;
  }

  const int hi16 = lane >> 4;

#if USE_ASYNC_X
  // Prefetch chunk 0 into buffer 0 before entering the recurrence.
  stage_x_async(x + (size_t)b0 * T_TOT, &s_x[wv][0][0], lane);
#endif

#pragma unroll 1
  for (int t = 0; t < T_TOT; ++t) {
    // ---- chunk boundary: uniform condition, wave-private buffers ----------
    if ((t & (XCHUNK - 1)) == 0) {
#if USE_ASYNC_X
      wait_asynccnt0();                     // current chunk's data has landed
      if (t + XCHUNK < T_TOT)               // prefetch next chunk, other buf
        stage_x_async(x + (size_t)b0 * T_TOT + t + XCHUNK,
                      &s_x[wv][((t >> 5) + 1) & 1][0], lane);
#else
      stage_x_sync(x + (size_t)b0 * T_TOT + t,
                   &s_x[wv][(t >> 5) & 1][0], lane);
#endif
    }
    const int buf = (t >> 5) & 1;
    const int tl  = t & (XCHUNK - 1);

    // x values for the 8 rows this lane accumulates (D-layout rows).
    v4f xlo = *(const v4f*)&s_x[wv][buf][tl * RPW + 8 * hi16];
    v4f xhi = *(const v4f*)&s_x[wv][buf][tl * RPW + 8 * hi16 + 4];

    // ------------------------------ layer 0 -------------------------------
#pragma unroll
    for (int hh = 0; hh < 2; ++hh) {
      v8f d[4];
#pragma unroll
      for (int gi = 0; gi < 4; ++gi) {            // i,f,g,o
        const int j = gi * 2 + hh;
        v8f acc;
#pragma unroll
        for (int r = 0; r < 8; ++r) {
          const float xv = (r < 4) ? xlo[r] : xhi[r - 4];
          acc[r] = __builtin_fmaf(xv, wih0s[j], bias0s[j]);
        }
        d[gi] = wmma16x16x32(h0A, w0[j], acc);    // h0(t-1) x w_hh0
      }
      v8f ig = v8_sig(d[0]);
      v8f fg = v8_sig(d[1]);
      v8f gg = v8_tanh(d[2]);
      v8f og = v8_sig(d[3]);
      c0[hh] = fg * c0[hh] + ig * gg;
      v8f hn = og * v8_tanh(c0[hh]);
      storeH(&s_h[wv][0][0], hn, hh, lane);
    }

    // D->A layout turn through LDS (same-wave LDS is in-order); this value
    // is used both for layer 1 now and as layer 0's recurrence next step.
    h0A = loadA(&s_h[wv][0][0], lane);

    // ------------------------------ layer 1 -------------------------------
#pragma unroll
    for (int hh = 0; hh < 2; ++hh) {
      v8f d[4];
#pragma unroll
      for (int gi = 0; gi < 4; ++gi) {
        const int j = gi * 2 + hh;
        v8f acc;
#pragma unroll
        for (int r = 0; r < 8; ++r) acc[r] = bias1s[j];
        acc   = wmma16x16x32(h0A, w1[j], acc);    // h0(t)   x w_ih1
        d[gi] = wmma16x16x32(h1A, w2[j], acc);    // h1(t-1) x w_hh1
      }
      v8f ig = v8_sig(d[0]);
      v8f fg = v8_sig(d[1]);
      v8f gg = v8_tanh(d[2]);
      v8f og = v8_sig(d[3]);
      c1[hh] = fg * c1[hh] + ig * gg;
      v8f hn = og * v8_tanh(c1[hh]);
      storeH(&s_h[wv][1][0], hn, hh, lane);
    }

    // Carry h1(t) as next step's A operand.
    h1A = loadA(&s_h[wv][1][0], lane);
  }

  // ------------------------- head: fc1 -> relu -> fc2 ---------------------
  if (lane < 16) {
    float hv[HID];
#pragma unroll
    for (int k = 0; k < HID; ++k)
      hv[k] = (float)s_h[wv][1][lane * HID + k];
    float o = fc2b[0];
#pragma unroll 4
    for (int u = 0; u < 16; ++u) {
      float a = fc1b[u];
#pragma unroll
      for (int k = 0; k < HID; ++k)
        a = __builtin_fmaf(fc1w[u * HID + k], hv[k], a);
      a = fmaxf(a, 0.0f);
      o = __builtin_fmaf(fc2w[u], a, o);
    }
    out[b0 + lane] = o;
  }
}

// ---------------------------------------------------------------------------
extern "C" void kernel_launch(void* const* d_in, const int* in_sizes, int n_in,
                              void* d_out, int out_size, void* d_ws,
                              size_t ws_size, hipStream_t stream) {
  (void)in_sizes; (void)n_in; (void)out_size; (void)ws_size;
  const float* x     = (const float*)d_in[0];
  const float* wih0  = (const float*)d_in[1];
  const float* whh0  = (const float*)d_in[2];
  const float* bih0  = (const float*)d_in[3];
  const float* bhh0  = (const float*)d_in[4];
  const float* wih1  = (const float*)d_in[5];
  const float* whh1  = (const float*)d_in[6];
  const float* bih1  = (const float*)d_in[7];
  const float* bhh1  = (const float*)d_in[8];
  const float* fc1w  = (const float*)d_in[9];
  const float* fc1b  = (const float*)d_in[10];
  const float* fc2w  = (const float*)d_in[11];
  const float* fc2b  = (const float*)d_in[12];
  _Float16* pk = (_Float16*)d_ws;   // 24 KB packed B-operand weights
  float* out = (float*)d_out;

  lstm_pack_weights<<<3, 256, 0, stream>>>(whh0, wih1, whh1, pk);

  const int blocks = B_TOT / (WPB * RPW);   // 64
  lstm_fused_kernel<<<blocks, WPB * 32, 0, stream>>>(
      x, wih0, bih0, bhh0, bih1, bhh1, fc1w, fc1b, fc2w, fc2b, pk, out);
}
